// VectorQuantizeEMA_9234179687620
// MI455X (gfx1250) — compile-verified, matrix operand order checked by reading, not run on hardware
//
#include <hip/hip_runtime.h>
#include <hip/hip_bf16.h>

typedef __attribute__((ext_vector_type(16))) __bf16 v16bf;
typedef __attribute__((ext_vector_type(8)))  float  v8f;
typedef __attribute__((ext_vector_type(4)))  int    v4i;
typedef unsigned short u16;

#define CDIM    256
#define NCODE   4096
#define HWROWS  1024          // rows per batch slab (H*W)
#define SLAB    (CDIM*HWROWS) // 262144 elements per batch in z_e
#define CHUNKS  (NCODE/16)    // 256 code chunks of 16
#define KSTEPS  (CDIM/32)     // 8 k-steps of 32

#if defined(__has_builtin)
#if __has_builtin(__builtin_amdgcn_global_load_async_to_lds_b128) && \
    __has_builtin(__builtin_amdgcn_s_wait_asynccnt)
#define HAVE_ASYNC_LDS 1
#endif
#endif

#if HAVE_ASYNC_LDS
typedef __attribute__((address_space(1))) v4i gv4i;
typedef __attribute__((address_space(3))) v4i lv4i;
#endif

// ---------------------------------------------------------------------------
// Kernel 1: convert codebook to bf16 hi/lo pair in WMMA B-fragment layout,
// and compute ||e||^2 per code.
// Fragment layout: [chunk][kstep][lane][16 ushorts]; lane l holds column
// N = chunk*16 + (l%16), K = kstep*32 + (l<16?0:8) + (j<8 ? j : j+8).
// ---------------------------------------------------------------------------
__global__ __launch_bounds__(256) void prep_embed(
    const float* __restrict__ embed,
    u16* __restrict__ bh, u16* __restrict__ bl,
    float* __restrict__ e2)
{
    __shared__ float ssq[256];
    int c = blockIdx.x;          // chunk
    int t = threadIdx.x;         // 0..255
    int l = t & 31;
    int s = t >> 5;
    int rs = l & 15;
    int row = c * 16 + rs;
    int kb = s * 32 + ((l < 16) ? 0 : 8);

    const float* src = embed + (size_t)row * CDIM + kb;
    float v[16];
#pragma unroll
    for (int j = 0; j < 8; ++j) v[j] = src[j];
#pragma unroll
    for (int j = 0; j < 8; ++j) v[8 + j] = src[16 + j];

    float sq = 0.f;
    u16 h[16], lo[16];
#pragma unroll
    for (int j = 0; j < 16; ++j) {
        float f = v[j];
        sq += f * f;
        __bf16 hb = (__bf16)f;
        float  r  = f - (float)hb;        // exact residual
        __bf16 lb = (__bf16)r;
        h[j]  = __builtin_bit_cast(u16, hb);
        lo[j] = __builtin_bit_cast(u16, lb);
    }

    size_t off = ((size_t)(c * KSTEPS + s) * 32 + l) * 16; // ushort index, 32B aligned
    uint4* dh = (uint4*)(bh + off);
    uint4* dl = (uint4*)(bl + off);
    dh[0] = ((uint4*)h)[0];  dh[1] = ((uint4*)h)[1];
    dl[0] = ((uint4*)lo)[0]; dl[1] = ((uint4*)lo)[1];

    // ||e||^2 : 16 threads share each row; fixed-order tree for determinism
    ssq[t] = sq;
    __syncthreads();
    if (t < 16) {
        float a = 0.f;
#pragma unroll
        for (int k = 0; k < 16; ++k) a += ssq[k * 16 + t];
        e2[c * 16 + t] = a;
    }
}

// ---------------------------------------------------------------------------
// Kernel 2: fused distance-GEMM (split-bf16 WMMA) + argmin + gather + ST out.
// Block = 8 waves; wave w owns rows [blk*128 + w*16, +16). Waves share B
// fragments through double-buffered LDS.
// ---------------------------------------------------------------------------
__global__ __launch_bounds__(256, 1) void vq_main(
    const float* __restrict__ ze, const float* __restrict__ embed,
    const u16* __restrict__ bh, const u16* __restrict__ bl,
    const float* __restrict__ e2, float* __restrict__ partial,
    float* __restrict__ zq_out, int* __restrict__ ind_out)
{
    __shared__ uint4  sB[2][2][512];   // [buf][hi/lo][frag data]  32 KB
    __shared__ float  sE2[NCODE];      // 16 KB
    __shared__ int    sIdx[128];

    int t = threadIdx.x;
    int l = t & 31;
    int w = t >> 5;
    int m = l & 15;
    int blk = blockIdx.x;

#pragma unroll
    for (int j = 0; j < 16; ++j) sE2[j * 256 + t] = e2[j * 256 + t];

    // ---- load + split A fragments for this wave's 16 rows, all K ----------
    int gRowBase = blk * 128 + w * 16;
    int b  = gRowBase >> 10;
    int sr = gRowBase & 1023;
    const float* aBase = ze + (size_t)b * SLAB + sr + m;
    int kb = (l < 16) ? 0 : 8;

    v16bf ahi[KSTEPS], alo[KSTEPS];
    float sumsq = 0.f;
#pragma unroll
    for (int s = 0; s < KSTEPS; ++s) {
#pragma unroll
        for (int j = 0; j < 16; ++j) {
            int K = s * 32 + kb + ((j < 8) ? j : j + 8);
            float f = aBase[(size_t)K * HWROWS];
            sumsq += f * f;
            __bf16 hb = (__bf16)f;
            float  r  = f - (float)hb;
            ahi[s][j] = hb;
            alo[s][j] = (__bf16)r;
        }
    }

    auto stage = [&](int chunk, int buf) {
        const uint4* srcH = (const uint4*)bh + (size_t)chunk * 512;
        const uint4* srcL = (const uint4*)bl + (size_t)chunk * 512;
#if HAVE_ASYNC_LDS
        __builtin_amdgcn_global_load_async_to_lds_b128((gv4i*)(srcH + t),       (lv4i*)&sB[buf][0][t],       0, 0);
        __builtin_amdgcn_global_load_async_to_lds_b128((gv4i*)(srcH + t + 256), (lv4i*)&sB[buf][0][t + 256], 0, 0);
        __builtin_amdgcn_global_load_async_to_lds_b128((gv4i*)(srcL + t),       (lv4i*)&sB[buf][1][t],       0, 0);
        __builtin_amdgcn_global_load_async_to_lds_b128((gv4i*)(srcL + t + 256), (lv4i*)&sB[buf][1][t + 256], 0, 0);
#else
        sB[buf][0][t]       = srcH[t];
        sB[buf][0][t + 256] = srcH[t + 256];
        sB[buf][1][t]       = srcL[t];
        sB[buf][1][t + 256] = srcL[t + 256];
#endif
    };

    float minv[8];
    int   mini[8];
#pragma unroll
    for (int i = 0; i < 8; ++i) { minv[i] = 3.4e38f; mini[i] = 0; }

    stage(0, 0);
    for (int chunk = 0; chunk < CHUNKS; ++chunk) {
        int buf = chunk & 1;
#if HAVE_ASYNC_LDS
        __builtin_amdgcn_s_wait_asynccnt(0);
#endif
        __syncthreads();
        if (chunk + 1 < CHUNKS) stage(chunk + 1, buf ^ 1);

        float e2v = sE2[chunk * 16 + m];
        v8f acc = {0.f, 0.f, 0.f, 0.f, 0.f, 0.f, 0.f, 0.f};
#pragma unroll
        for (int s = 0; s < KSTEPS; ++s) {
            union { uint4 u[2]; v16bf v; } BH, BL;
            int fi = (s * 32 + l) * 2;
            BH.u[0] = sB[buf][0][fi];  BH.u[1] = sB[buf][0][fi + 1];
            BL.u[0] = sB[buf][1][fi];  BL.u[1] = sB[buf][1][fi + 1];
            acc = __builtin_amdgcn_wmma_f32_16x16x32_bf16(false, ahi[s], false, BH.v, (short)0, acc, false, false);
            acc = __builtin_amdgcn_wmma_f32_16x16x32_bf16(false, ahi[s], false, BL.v, (short)0, acc, false, false);
            acc = __builtin_amdgcn_wmma_f32_16x16x32_bf16(false, alo[s], false, BH.v, (short)0, acc, false, false);
        }

        int n = chunk * 16 + m;   // this lane's candidate code index
#pragma unroll
        for (int i = 0; i < 8; ++i) {
            float d = e2v - 2.0f * acc[i];   // ||e||^2 - 2 x.e  (||x||^2 const per row)
            if (d < minv[i]) { minv[i] = d; mini[i] = n; }
        }
    }

    // ---- per-row argmin across the 16 lanes of each half-wave --------------
#pragma unroll
    for (int i = 0; i < 8; ++i) {
        float v = minv[i];
        int  id = mini[i];
#pragma unroll
        for (int o = 1; o < 16; o <<= 1) {
            float ov = __shfl_xor(v, o, 32);
            int   oi = __shfl_xor(id, o, 32);
            if (ov < v || (ov == v && oi < id)) { v = ov; id = oi; }
        }
        minv[i] = v; mini[i] = id;
    }

    // row sums of squares: rowSq(m) = sumsq(lane m) + sumsq(lane m+16)
    float s2  = sumsq + __shfl_xor(sumsq, 16, 32);
    float s2x = __shfl_xor(s2, 8, 32);

    float contrib = 0.f;
    if (m < 8) {
        int i = m;                           // accumulator slot this lane owns
        int rowLocal = ((l >> 4) << 3) + i;  // 0..15 within wave tile
        int gr = gRowBase + rowLocal;
        int id = mini[i];
        ind_out[gr] = id;
        sIdx[w * 16 + rowLocal] = id;
        float rowSq = (l < 16) ? s2 : s2x;
        contrib = rowSq + minv[i];           // ||x - e*||^2 for this row
    }
#pragma unroll
    for (int o = 1; o < 32; o <<= 1) contrib += __shfl_xor(contrib, o, 32);
    if (l == 0) partial[blk * 8 + w] = contrib;

    // ---- z_q straight-through output: out = ze + (q - ze), [B,C,H,W] -------
    __syncthreads();
    int r  = t & 127;
    int c0 = (t >> 7) * 128;
    int gr = blk * 128 + r;
    int bb = gr >> 10;
    int srow = gr & 1023;
    int idx = sIdx[r];
    const float* erow = embed + (size_t)idx * CDIM + c0;
    const float* zrow = ze + (size_t)bb * SLAB + srow;
    float* orow = zq_out + (size_t)bb * SLAB + srow;
#pragma unroll 4
    for (int c = 0; c < 128; ++c) {
        float q = erow[c];
        float z = zrow[(size_t)(c0 + c) * HWROWS];
        orow[(size_t)(c0 + c) * HWROWS] = z + (q - z);
    }
}

// ---------------------------------------------------------------------------
// Kernel 3: deterministic reduction of 2048 per-wave partials -> diff scalar
// ---------------------------------------------------------------------------
__global__ __launch_bounds__(256) void vq_reduce(
    const float* __restrict__ partial, float* __restrict__ diff_out)
{
    __shared__ float s[256];
    int t = threadIdx.x;
    float a = 0.f;
#pragma unroll
    for (int j = 0; j < 8; ++j) a += partial[j * 256 + t];
    s[t] = a;
    __syncthreads();
    for (int off = 128; off > 0; off >>= 1) {
        if (t < off) s[t] += s[t + off];
        __syncthreads();
    }
    if (t == 0) diff_out[0] = s[0] * (1.0f / 8388608.0f); // mean over 32*256*32*32
}

extern "C" void kernel_launch(void* const* d_in, const int* in_sizes, int n_in,
                              void* d_out, int out_size, void* d_ws, size_t ws_size,
                              hipStream_t stream)
{
    const float* ze    = (const float*)d_in[0];   // [32,256,32,32]
    const float* embed = (const float*)d_in[1];   // [4096,256]

    float* zq   = (float*)d_out;                  // [32,256,32,32]
    float* diff = (float*)d_out + 8388608;        // scalar
    int*   ind  = (int*)((float*)d_out + 8388609);// [32,32,32] int32

    char* ws = (char*)d_ws;
    u16*   bh      = (u16*)ws;                              // 2 MB
    u16*   bl      = (u16*)(ws + (2u << 20));               // 2 MB
    float* e2      = (float*)(ws + (4u << 20));             // 16 KB
    float* partial = (float*)(ws + (4u << 20) + (16u << 10)); // 8 KB

    prep_embed<<<CHUNKS, 256, 0, stream>>>(embed, bh, bl, e2);
    vq_main<<<256, 256, 0, stream>>>(ze, embed, bh, bl, e2, partial, zq, ind);
    vq_reduce<<<1, 256, 0, stream>>>(partial, diff);
}
